// BitMGQA_64647847739670
// MI455X (gfx1250) — 16-bit and 8-bit matrix operand order repaired and checked by reading, compiled, not run on hardware
//
#include <hip/hip_runtime.h>
#include <stdint.h>

#define EMBED 1024
#define KVE   256
#define BATCH 2
#define SEQ   2048
#define NTOK  (BATCH * SEQ)
#define DHEAD 64
#define GKV   4
#define NQH   16

typedef __attribute__((ext_vector_type(8)))  int      v8i;
typedef __attribute__((ext_vector_type(4)))  int      v4i;
typedef __attribute__((ext_vector_type(8)))  float    v8f;
typedef __attribute__((ext_vector_type(16))) _Float16 v16h;

union V16H { v16h h; v8i i; v4i q[2]; };

static __device__ inline v8f zero8f() {
  v8f z;
#pragma unroll
  for (int i = 0; i < 8; ++i) z[i] = 0.f;
  return z;
}

static __device__ inline v8i zero8i() {
  v8i z;
#pragma unroll
  for (int i = 0; i < 8; ++i) z[i] = 0;
  return z;
}

__global__ void k_zero(float* p, int n) {
  int i = blockIdx.x * blockDim.x + threadIdx.x;
  if (i < n) p[i] = 0.f;
}

__global__ __launch_bounds__(256) void k_abssum(const float* __restrict__ w, int n,
                                                float* __restrict__ dst) {
  __shared__ float red[256];
  float s = 0.f;
  for (int i = blockIdx.x * blockDim.x + threadIdx.x; i < n; i += gridDim.x * blockDim.x)
    s += fabsf(w[i]);
  red[threadIdx.x] = s; __syncthreads();
  for (int o = 128; o > 0; o >>= 1) {
    if (threadIdx.x < o) red[threadIdx.x] += red[threadIdx.x + o];
    __syncthreads();
  }
  if (threadIdx.x == 0) atomicAdd(dst, red[0]);
}

__global__ void k_sign(const float* __restrict__ w, int8_t* __restrict__ s, int n) {
  int i = blockIdx.x * blockDim.x + threadIdx.x;
  if (i < n) {
    float x = w[i];
    s[i] = (int8_t)((x > 0.f) - (x < 0.f));
  }
}

__global__ __launch_bounds__(256)
void k_quant_rows(const float* __restrict__ x, int8_t* __restrict__ xq,
                  float* __restrict__ rowscale) {
  __shared__ float red[256];
  const int row = blockIdx.x;
  const int tid = threadIdx.x;
  const float* xr = x + (size_t)row * EMBED;
  float v[4]; float ssq = 0.f;
#pragma unroll
  for (int j = 0; j < 4; ++j) { v[j] = xr[tid + j * 256]; ssq += v[j] * v[j]; }
  red[tid] = ssq; __syncthreads();
  for (int o = 128; o > 0; o >>= 1) {
    if (tid < o) red[tid] += red[tid + o];
    __syncthreads();
  }
  const float rinv = rsqrtf(red[0] * (1.f / EMBED) + 1e-6f) * (1.f / 32.f);
  float mx = 0.f;
#pragma unroll
  for (int j = 0; j < 4; ++j) { v[j] *= rinv; mx = fmaxf(mx, fabsf(v[j])); }
  __syncthreads();
  red[tid] = mx; __syncthreads();
  for (int o = 128; o > 0; o >>= 1) {
    if (tid < o) red[tid] = fmaxf(red[tid], red[tid + o]);
    __syncthreads();
  }
  const float scale = 127.f / fmaxf(red[0], 1e-5f);
#pragma unroll
  for (int j = 0; j < 4; ++j) {
    float q = rintf(v[j] * scale);
    q = fminf(fmaxf(q, -128.f), 127.f);
    xq[(size_t)row * EMBED + tid + j * 256] = (int8_t)q;
  }
  if (tid == 0) rowscale[row] = 1.f / scale;
}

__global__ __launch_bounds__(256)
void k_gemm_iu8(const int8_t* __restrict__ Xq, const float* __restrict__ rowscale,
                const int8_t* __restrict__ Wsgn, const float* __restrict__ bias,
                const float* __restrict__ alphaSum, float invCount,
                void* __restrict__ out, int N, int K, int mode) {
  const int lane = threadIdx.x & 31;
  const int wv   = threadIdx.x >> 5;
  const int m0   = blockIdx.y * 16;
  const int n0   = (blockIdx.x * 8 + wv) * 16;
  const int h    = lane >> 4;
  const int ml   = lane & 15;

  const int8_t* arow = Xq   + (size_t)(m0 + ml) * K;
  const int8_t* brow = Wsgn + (size_t)(n0 + ml) * K;

  v8i acc = zero8i();
  for (int k0 = 0; k0 < K; k0 += 64) {
    v8i a, bm;
#pragma unroll
    for (int g = 0; g < 4; ++g) {
      const int* p = (const int*)(arow + k0 + (g >> 1) * 32 + h * 16 + (g & 1) * 8);
      a[2 * g] = p[0]; a[2 * g + 1] = p[1];
    }
    const v4i b0 = *(const v4i*)(brow + k0 + h * 16);
    const v4i b1 = *(const v4i*)(brow + k0 + 32 + h * 16);
#pragma unroll
    for (int j = 0; j < 4; ++j) { bm[j] = b0[j]; bm[j + 4] = b1[j]; }
    acc = __builtin_amdgcn_wmma_i32_16x16x64_iu8(true, a, true, bm, acc, false, false);
  }

  const float alpha = alphaSum[0] * invCount;
#pragma unroll
  for (int r = 0; r < 8; ++r) {
    const int m = m0 + r + h * 8;
    const int n = n0 + ml;
    const float val = alpha * rowscale[m] * (float)acc[r] + bias[n];
    if (mode == 0) {
      ((float*)out)[(size_t)m * N + n] = val;
    } else if (mode == 1) {
      ((_Float16*)out)[(size_t)m * N + n] = (_Float16)val;
    } else {
      const int b = m >> 11, t = m & 2047;
      const int g = n >> 6,  d = n & 63;
      ((_Float16*)out)[(((size_t)(b * GKV + g) * DHEAD + d) * SEQ) + t] = (_Float16)val;
    }
  }
}

__global__ __launch_bounds__(128)
void k_attention(const _Float16* __restrict__ qh, const _Float16* __restrict__ kh,
                 const _Float16* __restrict__ vt, float* __restrict__ attn) {
  __shared__ _Float16 lds_p[4][16][32];
  const int lane = threadIdx.x & 31;
  const int wv   = threadIdx.x >> 5;
  const int W    = blockIdx.x * 4 + wv;
  const int qblk = W & 127;
  const int head = (W >> 7) & 15;
  const int b    = W >> 11;
  const int g    = head >> 2;
  const int s0   = qblk * 16;
  const int h    = lane >> 4;
  const int ml   = lane & 15;

  const _Float16* qrow = qh + ((size_t)(b * SEQ + s0 + ml) * EMBED + head * DHEAD);
  V16H aq0, aq1;
  aq0.q[0] = *(const v4i*)(qrow + h * 16);
  aq0.q[1] = *(const v4i*)(qrow + 8 + h * 16);
  aq1.q[0] = *(const v4i*)(qrow + 32 + h * 16);
  aq1.q[1] = *(const v4i*)(qrow + 40 + h * 16);

  v8f o[4]; float mrow[8], lrow[8];
#pragma unroll
  for (int a = 0; a < 4; ++a) o[a] = zero8f();
#pragma unroll
  for (int r = 0; r < 8; ++r) { mrow[r] = -3.0e38f; lrow[r] = 0.f; }

  const float sscale = 0.125f;

  for (int t0 = 0; t0 < SEQ; t0 += 32) {
    const _Float16* kr0 = kh + ((size_t)(b * SEQ + t0 + ml) * KVE + g * DHEAD);
    const _Float16* kr1 = kr0 + (size_t)16 * KVE;
    V16H bk;
    v8f c0 = zero8f(), c1 = zero8f();
    bk.q[0] = *(const v4i*)(kr0 + h * 16); bk.q[1] = *(const v4i*)(kr0 + h * 16 + 8);
    c0 = __builtin_amdgcn_wmma_f32_16x16x32_f16(false, aq0.h, false, bk.h, (short)0, c0, false, false);
    bk.q[0] = *(const v4i*)(kr0 + 32 + h * 16); bk.q[1] = *(const v4i*)(kr0 + 40 + h * 16);
    c0 = __builtin_amdgcn_wmma_f32_16x16x32_f16(false, aq1.h, false, bk.h, (short)0, c0, false, false);
    bk.q[0] = *(const v4i*)(kr1 + h * 16); bk.q[1] = *(const v4i*)(kr1 + h * 16 + 8);
    c1 = __builtin_amdgcn_wmma_f32_16x16x32_f16(false, aq0.h, false, bk.h, (short)0, c1, false, false);
    bk.q[0] = *(const v4i*)(kr1 + 32 + h * 16); bk.q[1] = *(const v4i*)(kr1 + 40 + h * 16);
    c1 = __builtin_amdgcn_wmma_f32_16x16x32_f16(false, aq1.h, false, bk.h, (short)0, c1, false, false);

#pragma unroll
    for (int r = 0; r < 8; ++r) {
      c0[r] *= sscale; c1[r] *= sscale;
      float vmax = fmaxf(c0[r], c1[r]);
#pragma unroll
      for (int off = 8; off >= 1; off >>= 1) vmax = fmaxf(vmax, __shfl_xor(vmax, off, 32));
      const float mnew = fmaxf(mrow[r], vmax);
      const float corr = __expf(mrow[r] - mnew);
      const float p0 = __expf(c0[r] - mnew);
      const float p1 = __expf(c1[r] - mnew);
      float ps = p0 + p1;
#pragma unroll
      for (int off = 8; off >= 1; off >>= 1) ps += __shfl_xor(ps, off, 32);
      lrow[r] = lrow[r] * corr + ps;
      mrow[r] = mnew;
#pragma unroll
      for (int a = 0; a < 4; ++a) o[a][r] *= corr;
      lds_p[wv][r + h * 8][ml]      = (_Float16)p0;
      lds_p[wv][r + h * 8][16 + ml] = (_Float16)p1;
    }

    V16H ap;
    const _Float16* pr = &lds_p[wv][ml][0];
    ap.q[0] = *(const v4i*)(pr + h * 16);
    ap.q[1] = *(const v4i*)(pr + 8 + h * 16);

    const _Float16* vbase = vt + ((size_t)(b * GKV + g) * DHEAD) * SEQ + t0 + h * 16;
#pragma unroll
    for (int a = 0; a < 4; ++a) {
      const _Float16* vr = vbase + (size_t)(a * 16 + ml) * SEQ;
      V16H bv;
      bv.q[0] = *(const v4i*)(vr);
      bv.q[1] = *(const v4i*)(vr + 8);
      o[a] = __builtin_amdgcn_wmma_f32_16x16x32_f16(false, ap.h, false, bv.h, (short)0, o[a], false, false);
    }
  }

#pragma unroll
  for (int a = 0; a < 4; ++a)
#pragma unroll
    for (int r = 0; r < 8; ++r) {
      const int m = r + h * 8;
      attn[(size_t)(b * SEQ + s0 + m) * EMBED + head * DHEAD + a * 16 + ml] =
          o[a][r] / lrow[r];
    }
}

extern "C" void kernel_launch(void* const* d_in, const int* in_sizes, int n_in,
                              void* d_out, int out_size, void* d_ws, size_t ws_size,
                              hipStream_t stream) {
  (void)in_sizes; (void)n_in; (void)out_size; (void)ws_size;
  const float* query = (const float*)d_in[0];
  const float* key_  = (const float*)d_in[1];
  const float* value = (const float*)d_in[2];
  const float* Wq = (const float*)d_in[3]; const float* bq = (const float*)d_in[4];
  const float* Wk = (const float*)d_in[5]; const float* bk = (const float*)d_in[6];
  const float* Wv = (const float*)d_in[7]; const float* bv = (const float*)d_in[8];
  const float* Wo = (const float*)d_in[9]; const float* bo = (const float*)d_in[10];
  float* out = (float*)d_out;

  uint8_t* ws = (uint8_t*)d_ws;
  size_t off = 0;
  auto alloc = [&](size_t bytes) -> void* {
    void* p = ws + off; off = (off + bytes + 255) & ~(size_t)255; return p;
  };
  float*    alphaSum = (float*)alloc(4 * sizeof(float));
  int8_t*   sWq = (int8_t*)alloc((size_t)EMBED * EMBED);
  int8_t*   sWk = (int8_t*)alloc((size_t)KVE * EMBED);
  int8_t*   sWv = (int8_t*)alloc((size_t)KVE * EMBED);
  int8_t*   sWo = (int8_t*)alloc((size_t)EMBED * EMBED);
  int8_t*   xq  = (int8_t*)alloc((size_t)NTOK * EMBED);
  float*    rs  = (float*)alloc((size_t)NTOK * sizeof(float));
  _Float16* qh  = (_Float16*)alloc((size_t)NTOK * EMBED * 2);
  _Float16* kh  = (_Float16*)alloc((size_t)NTOK * KVE * 2);
  _Float16* vt  = (_Float16*)alloc((size_t)NTOK * KVE * 2);
  float*    attn = (float*)alloc((size_t)NTOK * EMBED * sizeof(float));

  k_zero<<<1, 32, 0, stream>>>(alphaSum, 4);
  k_abssum<<<256, 256, 0, stream>>>(Wq, EMBED * EMBED, alphaSum + 0);
  k_abssum<<<256, 256, 0, stream>>>(Wk, KVE * EMBED,   alphaSum + 1);
  k_abssum<<<256, 256, 0, stream>>>(Wv, KVE * EMBED,   alphaSum + 2);
  k_abssum<<<256, 256, 0, stream>>>(Wo, EMBED * EMBED, alphaSum + 3);
  k_sign<<<(EMBED * EMBED) / 256, 256, 0, stream>>>(Wq, sWq, EMBED * EMBED);
  k_sign<<<(KVE * EMBED) / 256,   256, 0, stream>>>(Wk, sWk, KVE * EMBED);
  k_sign<<<(KVE * EMBED) / 256,   256, 0, stream>>>(Wv, sWv, KVE * EMBED);
  k_sign<<<(EMBED * EMBED) / 256, 256, 0, stream>>>(Wo, sWo, EMBED * EMBED);

  const dim3 gq(EMBED / 128, NTOK / 16);
  const dim3 gk(KVE / 128,   NTOK / 16);
  const float invE  = 1.f / ((float)EMBED * (float)EMBED);
  const float invKV = 1.f / ((float)KVE * (float)EMBED);

  k_quant_rows<<<NTOK, 256, 0, stream>>>(query, xq, rs);
  k_gemm_iu8<<<gq, 256, 0, stream>>>(xq, rs, sWq, bq, alphaSum + 0, invE, qh, EMBED, EMBED, 1);
  k_quant_rows<<<NTOK, 256, 0, stream>>>(key_, xq, rs);
  k_gemm_iu8<<<gk, 256, 0, stream>>>(xq, rs, sWk, bk, alphaSum + 1, invKV, kh, KVE, EMBED, 1);
  k_quant_rows<<<NTOK, 256, 0, stream>>>(value, xq, rs);
  k_gemm_iu8<<<gk, 256, 0, stream>>>(xq, rs, sWv, bv, alphaSum + 2, invKV, vt, KVE, EMBED, 2);

  k_attention<<<(BATCH * NQH * (SEQ / 16)) / 4, 128, 0, stream>>>(qh, kh, vt, attn);

  k_quant_rows<<<NTOK, 256, 0, stream>>>(attn, xq, rs);
  k_gemm_iu8<<<gq, 256, 0, stream>>>(xq, rs, sWo, bo, alphaSum + 3, invE, out, EMBED, EMBED, 0);
}
